// FlashAttention_20916490732005
// MI455X (gfx1250) — matrix operand order repaired and checked by reading, not compiled, not run on hardware
//
#include <hip/hip_runtime.h>
#include <hip/hip_bf16.h>

// ---------------------------------------------------------------------------
// MI455X (gfx1250) fused attention layer.
// Compute-bound (~70 GFLOP vs ~50MB HBM) -> everything through
// v_wmma_f32_16x16x32_bf16 (fp32 accumulate). wave32, 4 waves / block.
// Each wave owns a 32-row strip (2 A operands) so every B operand fetched
// from LDS feeds two WMMAs.
// ---------------------------------------------------------------------------

typedef __bf16 bf16;
typedef __attribute__((ext_vector_type(16))) __bf16 v16bf;
typedef __attribute__((ext_vector_type(8)))  __bf16 v8bf;
typedef __attribute__((ext_vector_type(8)))  float  v8f;

#define LSTR 72   // LDS row stride in bf16 elements (144B: 16B-aligned, conflict-skewed)

// ---- WMMA operand gathers from LDS (CDNA5 ISA 7.12.2 layouts) -------------
// A (16x32 bf16, MxK): lanes 0-15: M=lane, K in {0..7, 16..23};
//                      lanes 16-31: M=lane-16, K in {8..15, 24..31}.
__device__ __forceinline__ v16bf lds_load_A(const bf16* base, int rowBase, int koff, int lane) {
    int m    = lane & 15;
    int ksel = (lane & 16) ? 8 : 0;
    const bf16* p = base + (rowBase + m) * LSTR + koff + ksel;
    v8bf c0 = *(const v8bf*)(p);        // K = koff+ksel+0..7
    v8bf c1 = *(const v8bf*)(p + 16);   // K = koff+ksel+16..23
    v16bf a;
#pragma unroll
    for (int i = 0; i < 8; ++i) { a[i] = c0[i]; a[8 + i] = c1[i]; }
    return a;
}

// B (32x16 bf16, KxN), stored in LDS transposed as Bt[n][k] (K contiguous):
// lanes 0-15: N=lane, K=koff+{0..7,16..23} ; lanes 16-31: N=lane-16, K=koff+{8..15,24..31}.
__device__ __forceinline__ v16bf lds_load_B(const bf16* base, int nBase, int koff, int lane) {
    int n  = lane & 15;
    int ks = koff + ((lane & 16) ? 8 : 0);
    const bf16* p = base + (nBase + n) * LSTR + ks;
    v8bf c0 = *(const v8bf*)(p);
    v8bf c1 = *(const v8bf*)(p + 16);
    v16bf b;
#pragma unroll
    for (int i = 0; i < 8; ++i) { b[i] = c0[i]; b[8 + i] = c1[i]; }
    return b;
}

__device__ __forceinline__ v8f wmma_bf16(v16bf a, v16bf b, v8f c) {
    return __builtin_amdgcn_wmma_f32_16x16x32_bf16(false, a, false, b, (short)0, c, false, false);
}

__device__ __forceinline__ v8f vzero() {
    return (v8f){0.f,0.f,0.f,0.f,0.f,0.f,0.f,0.f};
}

// ---------------------------------------------------------------------------
__global__ void __launch_bounds__(256) f32_to_bf16_k(const float* __restrict__ src,
                                                     bf16* __restrict__ dst, int n) {
    int i = blockIdx.x * 256 + threadIdx.x;
    if (i < n) dst[i] = (bf16)src[i];
}

// ---------------------------------------------------------------------------
// QKV projection: 128x64 output tile of q/k/v = x(4096x1024) @ W(1024x1024).
// grid = (32 row tiles, 16 heads, 3 {q,k,v}); block = 128 (4 waves x 32 rows).
// Output scattered to (B,H,S,D) bf16; scale 1/sqrt(D) folded into Q.
// ---------------------------------------------------------------------------
__global__ void __launch_bounds__(128) qkv_gemm_k(const bf16* __restrict__ xb,
                                                  const bf16* __restrict__ wqb,
                                                  const bf16* __restrict__ wkb,
                                                  const bf16* __restrict__ wvb,
                                                  bf16* __restrict__ qb,
                                                  bf16* __restrict__ kb,
                                                  bf16* __restrict__ vb) {
    __shared__ bf16 xs[128 * LSTR];
    __shared__ bf16 ws[64 * LSTR];   // W tile transposed: ws[n][k]
    const int tid  = threadIdx.x;
    const int lane = tid & 31;
    const int wave = tid >> 5;
    const int row0 = blockIdx.x * 128;
    const int h    = blockIdx.y;
    const int sel  = blockIdx.z;

    const bf16* W   = (sel == 0) ? wqb : (sel == 1) ? wkb : wvb;
    bf16*       out = (sel == 0) ? qb  : (sel == 1) ? kb  : vb;
    const float scale = (sel == 0) ? 0.125f : 1.0f;   // D^-0.5 folded into Q

    v8f acc[2][4];
#pragma unroll
    for (int mi = 0; mi < 2; ++mi)
#pragma unroll
        for (int t = 0; t < 4; ++t) acc[mi][t] = vzero();

    for (int k0 = 0; k0 < 1024; k0 += 64) {
        __syncthreads();
        if (k0 + 64 < 1024) {   // hide next tiles' HBM latency behind this tile
            __builtin_prefetch(&xb[(size_t)(row0 + tid) * 1024 + k0 + 64], 0, 1);
            __builtin_prefetch(&W[(size_t)(k0 + 64 + (tid & 63)) * 1024 + h * 64], 0, 1);
        }
        // x tile: 128 rows x 64 K, 8B chunks, coalesced
#pragma unroll
        for (int c = tid; c < 128 * 16; c += 128) {
            int r = c >> 4, off = (c & 15) << 2;
            *(uint2*)&xs[r * LSTR + off] =
                *(const uint2*)&xb[(size_t)(row0 + r) * 1024 + k0 + off];
        }
        // W tile transposed: 8B coalesced global loads, 4x b16 LDS scatter
#pragma unroll
        for (int c = tid; c < 1024; c += 128) {
            int kk = c >> 4, n4 = (c & 15) << 2;
            uint2 wv_ = *(const uint2*)&W[(size_t)(k0 + kk) * 1024 + h * 64 + n4];
            const bf16* wp = (const bf16*)&wv_;
            ws[(n4 + 0) * LSTR + kk] = wp[0];
            ws[(n4 + 1) * LSTR + kk] = wp[1];
            ws[(n4 + 2) * LSTR + kk] = wp[2];
            ws[(n4 + 3) * LSTR + kk] = wp[3];
        }
        __syncthreads();
#pragma unroll
        for (int kk2 = 0; kk2 < 64; kk2 += 32) {
            v16bf a0 = lds_load_A(xs, wave * 32,      kk2, lane);
            v16bf a1 = lds_load_A(xs, wave * 32 + 16, kk2, lane);
#pragma unroll
            for (int t = 0; t < 4; ++t) {
                v16bf b = lds_load_B(ws, t * 16, kk2, lane);
                acc[0][t] = wmma_bf16(a0, b, acc[0][t]);
                acc[1][t] = wmma_bf16(a1, b, acc[1][t]);
            }
        }
    }
    // C/D layout: lane<16 -> rows +0..7; lane>=16 -> rows +8..15
    const int rsel = (lane & 16) ? 8 : 0;
    const int ncol = lane & 15;
#pragma unroll
    for (int mi = 0; mi < 2; ++mi)
#pragma unroll
        for (int t = 0; t < 4; ++t)
#pragma unroll
            for (int r = 0; r < 8; ++r) {
                int rg = row0 + wave * 32 + mi * 16 + rsel + r;
                int bb = rg >> 11, s = rg & 2047;
                out[(((size_t)(bb * 16 + h)) * 2048 + s) * 64 + t * 16 + ncol] =
                    (bf16)(acc[mi][t][r] * scale);
            }
}

// ---------------------------------------------------------------------------
// Flash attention, faithful to reference (incl. per-iteration divide by l_new
// WITHOUT rescaling o by l_prev). Two reference q-blocks (128 rows) per
// workgroup -- legal since the online recurrence is purely per-row.
// grid = (16 q-tiles, 16 heads, 2 batch); 4 waves x 32 rows.
// ---------------------------------------------------------------------------
__global__ void __launch_bounds__(128) flash_attn_k(const bf16* __restrict__ qg,
                                                    const bf16* __restrict__ kg,
                                                    const bf16* __restrict__ vg,
                                                    bf16* __restrict__ attnb) {
    __shared__ bf16 qsm[128 * LSTR];  // Q tile  [row][d]
    __shared__ bf16 ksm[64 * LSTR];   // K tile  [kv][d]   (= B operand for Q.K^T)
    __shared__ bf16 vtm[64 * LSTR];   // V tile transposed [d][kv] (= B operand for P.V)
    __shared__ bf16 psm[128 * LSTR];  // P tile  [row][kv] (C-layout -> A-layout relay)

    const int tid  = threadIdx.x;
    const int lane = tid & 31;
    const int wave = tid >> 5;
    const int bh   = blockIdx.z * 16 + blockIdx.y;     // b*H + h
    const int q0   = blockIdx.x * 128;

    const bf16* qh = qg + (size_t)bh * 2048 * 64;
    const bf16* kh = kg + (size_t)bh * 2048 * 64;
    const bf16* vh = vg + (size_t)bh * 2048 * 64;

    // load Q tile once (128 rows x 64 d)
#pragma unroll
    for (int c = tid; c < 128 * 16; c += 128) {
        int r = c >> 4, off = (c & 15) << 2;
        *(uint2*)&qsm[r * LSTR + off] = *(const uint2*)&qh[(size_t)(q0 + r) * 64 + off];
    }

    v8f o[2][4];
    float mrow[2][8], lrow[2][8];
#pragma unroll
    for (int mi = 0; mi < 2; ++mi) {
#pragma unroll
        for (int t = 0; t < 4; ++t) o[mi][t] = vzero();
#pragma unroll
        for (int r = 0; r < 8; ++r) { mrow[mi][r] = -__builtin_inff(); lrow[mi][r] = 0.f; }
    }

    for (int j = 0; j < 32; ++j) {
        __syncthreads();   // protect previous iteration's K/V reads
        const bf16* kblk = kh + (size_t)j * 64 * 64;
        const bf16* vblk = vh + (size_t)j * 64 * 64;
        if (j + 1 < 32) {  // prefetch next K/V blocks (8KB each = 128 x 64B)
            __builtin_prefetch(kblk + 64 * 64 + tid * 32, 0, 1);
            __builtin_prefetch(vblk + 64 * 64 + tid * 32, 0, 1);
        }
#pragma unroll
        for (int c = tid; c < 64 * 16; c += 128) {
            int r = c >> 4, off = (c & 15) << 2;
            *(uint2*)&ksm[r * LSTR + off] = *(const uint2*)&kblk[r * 64 + off];
        }
#pragma unroll
        for (int c = tid; c < 1024; c += 128) {     // V transpose: 8B loads, b16 scatter
            int kv = c >> 4, d4 = (c & 15) << 2;
            uint2 vv = *(const uint2*)&vblk[kv * 64 + d4];
            const bf16* vp = (const bf16*)&vv;
            vtm[(d4 + 0) * LSTR + kv] = vp[0];
            vtm[(d4 + 1) * LSTR + kv] = vp[1];
            vtm[(d4 + 2) * LSTR + kv] = vp[2];
            vtm[(d4 + 3) * LSTR + kv] = vp[3];
        }
        __syncthreads();

        // S = Q . K^T   (scale pre-folded into Q)
        v8f s[2][4];
#pragma unroll
        for (int mi = 0; mi < 2; ++mi)
#pragma unroll
            for (int t = 0; t < 4; ++t) s[mi][t] = vzero();
#pragma unroll
        for (int kk2 = 0; kk2 < 64; kk2 += 32) {
            v16bf a0 = lds_load_A(qsm, wave * 32,      kk2, lane);
            v16bf a1 = lds_load_A(qsm, wave * 32 + 16, kk2, lane);
#pragma unroll
            for (int t = 0; t < 4; ++t) {
                v16bf b = lds_load_B(ksm, t * 16, kk2, lane);
                s[0][t] = wmma_bf16(a0, b, s[0][t]);
                s[1][t] = wmma_bf16(a1, b, s[1][t]);
            }
        }

        // online softmax; element r of each lane is one row, 16 lanes share it
        float mnew[2][8], alpha[2][8], lnew[2][8];
#pragma unroll
        for (int mi = 0; mi < 2; ++mi) {
#pragma unroll
            for (int r = 0; r < 8; ++r) {
                float mx = fmaxf(fmaxf(s[mi][0][r], s[mi][1][r]),
                                 fmaxf(s[mi][2][r], s[mi][3][r]));
                mx = fmaxf(mx, __shfl_xor(mx, 1));
                mx = fmaxf(mx, __shfl_xor(mx, 2));
                mx = fmaxf(mx, __shfl_xor(mx, 4));
                mx = fmaxf(mx, __shfl_xor(mx, 8));   // stays inside 16-lane half
                mnew[mi][r]  = fmaxf(mrow[mi][r], mx);
                alpha[mi][r] = __expf(mrow[mi][r] - mnew[mi][r]);
            }
#pragma unroll
            for (int t = 0; t < 4; ++t)
#pragma unroll
                for (int r = 0; r < 8; ++r)
                    s[mi][t][r] = __expf(s[mi][t][r] - mnew[mi][r]);
#pragma unroll
            for (int r = 0; r < 8; ++r) {
                float sm = s[mi][0][r] + s[mi][1][r] + s[mi][2][r] + s[mi][3][r];
                sm += __shfl_xor(sm, 1);
                sm += __shfl_xor(sm, 2);
                sm += __shfl_xor(sm, 4);
                sm += __shfl_xor(sm, 8);
                lnew[mi][r] = alpha[mi][r] * lrow[mi][r] + sm;
            }
        }

        // relay P through LDS: C-layout -> A-layout (wave-private rows)
        const int rsel = (lane & 16) ? 8 : 0;
        const int ncol = lane & 15;
#pragma unroll
        for (int mi = 0; mi < 2; ++mi)
#pragma unroll
            for (int t = 0; t < 4; ++t)
#pragma unroll
                for (int r = 0; r < 8; ++r)
                    psm[(wave * 32 + mi * 16 + rsel + r) * LSTR + t * 16 + ncol] =
                        (bf16)s[mi][t][r];

        // PV = P . V
        v8f pv[2][4];
#pragma unroll
        for (int mi = 0; mi < 2; ++mi)
#pragma unroll
            for (int t = 0; t < 4; ++t) pv[mi][t] = vzero();
#pragma unroll
        for (int kk2 = 0; kk2 < 64; kk2 += 32) {
            v16bf a0 = lds_load_A(psm, wave * 32,      kk2, lane);
            v16bf a1 = lds_load_A(psm, wave * 32 + 16, kk2, lane);
#pragma unroll
            for (int t = 0; t < 4; ++t) {
                v16bf b = lds_load_B(vtm, t * 16, kk2, lane);
                pv[0][t] = wmma_bf16(a0, b, pv[0][t]);
                pv[1][t] = wmma_bf16(a1, b, pv[1][t]);
            }
        }

        // o = (alpha*o + P.V) / l_new   (exactly as the reference does)
#pragma unroll
        for (int mi = 0; mi < 2; ++mi) {
#pragma unroll
            for (int t = 0; t < 4; ++t)
#pragma unroll
                for (int r = 0; r < 8; ++r)
                    o[mi][t][r] = (alpha[mi][r] * o[mi][t][r] + pv[mi][t][r]) / lnew[mi][r];
#pragma unroll
            for (int r = 0; r < 8; ++r) { mrow[mi][r] = mnew[mi][r]; lrow[mi][r] = lnew[mi][r]; }
        }
    }

    // store strips to attn buffer in (B, S, H*D) bf16
    const int bb   = bh >> 4, h = bh & 15;
    const int rsel = (lane & 16) ? 8 : 0;
    const int ncol = lane & 15;
#pragma unroll
    for (int mi = 0; mi < 2; ++mi)
#pragma unroll
        for (int t = 0; t < 4; ++t)
#pragma unroll
            for (int r = 0; r < 8; ++r) {
                int srow = q0 + wave * 32 + mi * 16 + rsel + r;
                attnb[((size_t)bb * 2048 + srow) * 1024 + h * 64 + t * 16 + ncol] =
                    (bf16)o[mi][t][r];
            }
}

// ---------------------------------------------------------------------------
// Output projection: out(4096x1024 f32) = attn(4096x1024 bf16) @ Wo
// grid = (32 row tiles, 16 col tiles), block = 128 (4 waves x 32 rows).
// ---------------------------------------------------------------------------
__global__ void __launch_bounds__(128) oproj_gemm_k(const bf16* __restrict__ ab,
                                                    const bf16* __restrict__ wob,
                                                    float* __restrict__ out) {
    __shared__ bf16 xs[128 * LSTR];
    __shared__ bf16 ws[64 * LSTR];
    const int tid  = threadIdx.x;
    const int lane = tid & 31;
    const int wave = tid >> 5;
    const int row0 = blockIdx.x * 128;
    const int col0 = blockIdx.y * 64;

    v8f acc[2][4];
#pragma unroll
    for (int mi = 0; mi < 2; ++mi)
#pragma unroll
        for (int t = 0; t < 4; ++t) acc[mi][t] = vzero();

    for (int k0 = 0; k0 < 1024; k0 += 64) {
        __syncthreads();
        if (k0 + 64 < 1024) {
            __builtin_prefetch(&ab[(size_t)(row0 + tid) * 1024 + k0 + 64], 0, 1);
            __builtin_prefetch(&wob[(size_t)(k0 + 64 + (tid & 63)) * 1024 + col0], 0, 1);
        }
#pragma unroll
        for (int c = tid; c < 128 * 16; c += 128) {
            int r = c >> 4, off = (c & 15) << 2;
            *(uint2*)&xs[r * LSTR + off] =
                *(const uint2*)&ab[(size_t)(row0 + r) * 1024 + k0 + off];
        }
#pragma unroll
        for (int c = tid; c < 1024; c += 128) {
            int kk = c >> 4, n4 = (c & 15) << 2;
            uint2 wv_ = *(const uint2*)&wob[(size_t)(k0 + kk) * 1024 + col0 + n4];
            const bf16* wp = (const bf16*)&wv_;
            ws[(n4 + 0) * LSTR + kk] = wp[0];
            ws[(n4 + 1) * LSTR + kk] = wp[1];
            ws[(n4 + 2) * LSTR + kk] = wp[2];
            ws[(n4 + 3) * LSTR + kk] = wp[3];
        }
        __syncthreads();
#pragma unroll
        for (int kk2 = 0; kk2 < 64; kk2 += 32) {
            v16bf a0 = lds_load_A(xs, wave * 32,      kk2, lane);
            v16bf a1 = lds_load_A(xs, wave * 32 + 16, kk2, lane);
#pragma unroll
            for (int t = 0; t < 4; ++t) {
                v16bf b = lds_load_B(ws, t * 16, kk2, lane);
                acc[0][t] = wmma_bf16(a0, b, acc[0][t]);
                acc[1][t] = wmma_bf16(a1, b, acc[1][t]);
            }
        }
    }
    const int rsel = (lane & 16) ? 8 : 0;
    const int ncol = lane & 15;
#pragma unroll
    for (int mi = 0; mi < 2; ++mi)
#pragma unroll
        for (int t = 0; t < 4; ++t)
#pragma unroll
            for (int r = 0; r < 8; ++r)
                out[(size_t)(row0 + wave * 32 + mi * 16 + rsel + r) * 1024 +
                    col0 + t * 16 + ncol] = acc[mi][t][r];
}

// ---------------------------------------------------------------------------
extern "C" void kernel_launch(void* const* d_in, const int* in_sizes, int n_in,
                              void* d_out, int out_size, void* d_ws, size_t ws_size,
                              hipStream_t stream) {
    (void)in_sizes; (void)n_in; (void)out_size; (void)ws_size;
    const float* x  = (const float*)d_in[0];
    const float* Wq = (const float*)d_in[1];
    const float* Wk = (const float*)d_in[2];
    const float* Wv = (const float*)d_in[3];
    const float* Wo = (const float*)d_in[4];
    float* out = (float*)d_out;

    char* ws = (char*)d_ws;
    size_t off = 0;
    const size_t XN = (size_t)4096 * 1024;     // x / attn elements
    const size_t WN = (size_t)1024 * 1024;     // weight elements
    const size_t QN = (size_t)2 * 16 * 2048 * 64;
    bf16* xb    = (bf16*)(ws + off); off += XN * 2;
    bf16* wqb   = (bf16*)(ws + off); off += WN * 2;
    bf16* wkb   = (bf16*)(ws + off); off += WN * 2;
    bf16* wvb   = (bf16*)(ws + off); off += WN * 2;
    bf16* wob   = (bf16*)(ws + off); off += WN * 2;
    bf16* qb    = (bf16*)(ws + off); off += QN * 2;
    bf16* kb    = (bf16*)(ws + off); off += QN * 2;
    bf16* vb    = (bf16*)(ws + off); off += QN * 2;
    bf16* attnb = (bf16*)(ws + off); off += XN * 2;   // ~48 MiB total

    f32_to_bf16_k<<<(int)(XN / 256), 256, 0, stream>>>(x,  xb,  (int)XN);
    f32_to_bf16_k<<<(int)(WN / 256), 256, 0, stream>>>(Wq, wqb, (int)WN);
    f32_to_bf16_k<<<(int)(WN / 256), 256, 0, stream>>>(Wk, wkb, (int)WN);
    f32_to_bf16_k<<<(int)(WN / 256), 256, 0, stream>>>(Wv, wvb, (int)WN);
    f32_to_bf16_k<<<(int)(WN / 256), 256, 0, stream>>>(Wo, wob, (int)WN);

    qkv_gemm_k<<<dim3(32, 16, 3), 128, 0, stream>>>(xb, wqb, wkb, wvb, qb, kb, vb);
    flash_attn_k<<<dim3(16, 16, 2), 128, 0, stream>>>(qb, kb, vb, attnb);
    oproj_gemm_k<<<dim3(32, 16), 128, 0, stream>>>(attnb, wob, out);
}
